// DynamicGATModel_47330539602432
// MI455X (gfx1250) — matrix operand order repaired and checked by reading, not compiled, not run on hardware
//
#include <hip/hip_runtime.h>
#include <math.h>

typedef __attribute__((ext_vector_type(16))) _Float16 v16h;
typedef __attribute__((ext_vector_type(8)))  float    v8f;

// ---------------------------------------------------------------------------
// Fill kernels (workspace init; graph-safe, rerun every launch)
// ---------------------------------------------------------------------------
__global__ void k_fill_f32(float* p, float v, long n) {
  long i = (long)blockIdx.x * blockDim.x + threadIdx.x;
  if (i < n) p[i] = v;
}
__global__ void k_fill_i32(int* p, int v, long n) {
  long i = (long)blockIdx.x * blockDim.x + threadIdx.x;
  if (i < n) p[i] = v;
}

// ---------------------------------------------------------------------------
// TimeEncode + concat:  Xin[n, 0:128] = node_features, Xin[n,128:256] = cos(..)
// ---------------------------------------------------------------------------
__global__ void k_time_merge(const float* __restrict__ nf, const float* __restrict__ ts,
                             const float* __restrict__ freq, const float* __restrict__ phase,
                             float* __restrict__ Xin, int N) {
  long i = (long)blockIdx.x * blockDim.x + threadIdx.x;
  if (i >= (long)N * 256) return;
  int n = (int)(i >> 8), c = (int)(i & 255);
  float v;
  if (c < 128) v = nf[(long)n * 128 + c];
  else { int d = c - 128; v = cosf(ts[n] * freq[d] + phase[d]); }
  Xin[i] = v;
}

// ---------------------------------------------------------------------------
// Pack weight matrix B[K,M] (f32, row-major) into WMMA B-fragment-major f16:
// fragment (kt, ct) = 32 lanes x 16 halves contiguous.
// Lane l: col = ct*16 + (l&15), h = l>>4; half 2v,2v+1 = B[(kt*32+2v+8h+(v>=4?8:0))*M+col], +M.
// ---------------------------------------------------------------------------
__global__ void k_pack_b(const float* __restrict__ B, _Float16* __restrict__ Bp,
                         int K, int M) {
  int tid = blockIdx.x * blockDim.x + threadIdx.x;
  int nct = M >> 4;
  int total = (K >> 5) * nct * 32;
  if (tid >= total) return;
  int lane = tid & 31;
  int ct = (tid >> 5) % nct;
  int kt = (tid >> 5) / nct;
  int col = ct * 16 + (lane & 15);
  int h = lane >> 4;
  int k0 = kt * 32;
  _Float16* o = Bp + (size_t)tid * 16;
  #pragma unroll
  for (int v = 0; v < 8; v++) {
    int k = k0 + 2 * v + 8 * h + (v >= 4 ? 8 : 0);
    o[2 * v]     = (_Float16)B[(size_t)k * M + col];
    o[2 * v + 1] = (_Float16)B[(size_t)(k + 1) * M + col];
  }
}

// ---------------------------------------------------------------------------
// WMMA GEMM: C[N,M] = A[N,K] * Bp (+ bias[M]).  A f32 via float4 loads, B pre-
// packed f16 fragments (one v16h load per tile). One wave = 16 rows x 64 cols.
// A-frag (16-bit 16x32): lane l: row=l&15, h=l>>4; halves = A[k0+8h .. +7],
// A[k0+16+8h .. +7].  C/D: VGPR r: M=r+8h, N=l&15.
// ---------------------------------------------------------------------------
template <int K>
__global__ void k_wmma_gemm(const float* __restrict__ A, const _Float16* __restrict__ Bp,
                            const float* __restrict__ bias, float* __restrict__ C,
                            int M, int rowTiles) {
  int wave = threadIdx.x >> 5;
  int lane = threadIdx.x & 31;
  int rt = blockIdx.y * 4 + wave;           // 4 waves / block, one row-tile each
  if (rt >= rowTiles) return;               // wave-uniform: EXEC stays all-ones
  int row0 = rt * 16;
  int ct0 = blockIdx.x * 4;                 // first of 4 col-tiles
  int nct = M >> 4;
  int r = lane & 15, h = lane >> 4;

  v8f acc[4];
  #pragma unroll
  for (int t = 0; t < 4; t++) acc[t] = (v8f){0.f,0.f,0.f,0.f,0.f,0.f,0.f,0.f};

  const float* Arow = A + (size_t)(row0 + r) * K + 8 * h;
  #pragma unroll
  for (int kt = 0; kt < K / 32; kt++) {
    int k0 = kt * 32;
    float4 a0 = *(const float4*)(Arow + k0);
    float4 a1 = *(const float4*)(Arow + k0 + 4);
    float4 a2 = *(const float4*)(Arow + k0 + 16);
    float4 a3 = *(const float4*)(Arow + k0 + 20);
    v16h a;
    a[0]  = (_Float16)a0.x; a[1]  = (_Float16)a0.y;
    a[2]  = (_Float16)a0.z; a[3]  = (_Float16)a0.w;
    a[4]  = (_Float16)a1.x; a[5]  = (_Float16)a1.y;
    a[6]  = (_Float16)a1.z; a[7]  = (_Float16)a1.w;
    a[8]  = (_Float16)a2.x; a[9]  = (_Float16)a2.y;
    a[10] = (_Float16)a2.z; a[11] = (_Float16)a2.w;
    a[12] = (_Float16)a3.x; a[13] = (_Float16)a3.y;
    a[14] = (_Float16)a3.z; a[15] = (_Float16)a3.w;
    #pragma unroll
    for (int t = 0; t < 4; t++) {
      v16h b = *(const v16h*)(Bp + ((size_t)(kt * nct + ct0 + t) * 32 + lane) * 16);
      acc[t] = __builtin_amdgcn_wmma_f32_16x16x32_f16(
          false, a, false, b, (short)0, acc[t], false, false);
    }
  }
  #pragma unroll
  for (int t = 0; t < 4; t++) {
    int col = (ct0 + t) * 16 + r;
    float bb = bias ? bias[col] : 0.f;
    #pragma unroll
    for (int i = 0; i < 8; i++) {
      int m = i + 8 * h;
      C[(size_t)(row0 + m) * M + col] = acc[t][i] + bb;
    }
  }
}

// ---------------------------------------------------------------------------
// Per-node attention coefficients: asrc[n]=dot(xp[n],a_src), adst likewise.
// ---------------------------------------------------------------------------
__global__ void k_attn(const float* __restrict__ xp, const float* __restrict__ aw,
                       const float* __restrict__ dw, float* __restrict__ asrc,
                       float* __restrict__ adst, int N) {
  long gid = (long)blockIdx.x * blockDim.x + threadIdx.x;
  int node = (int)(gid >> 5), lane = (int)(gid & 31);
  if (node >= N) return;
  float4 v = ((const float4*)(xp + (size_t)node * 128))[lane];
  float4 a = ((const float4*)aw)[lane];
  float4 d = ((const float4*)dw)[lane];
  float ps = v.x * a.x + v.y * a.y + v.z * a.z + v.w * a.w;
  float pd = v.x * d.x + v.y * d.y + v.z * d.z + v.w * d.w;
  for (int off = 16; off; off >>= 1) {
    ps += __shfl_down(ps, off, 32);
    pd += __shfl_down(pd, off, 32);
  }
  if (lane == 0) { asrc[node] = ps; adst[node] = pd; }
}

// ---------------------------------------------------------------------------
// Edge segment-softmax: ordered-int float atomicMax trick
// ---------------------------------------------------------------------------
__device__ __forceinline__ int   f2ord(float f) { int i = __float_as_int(f); return i >= 0 ? i : (i ^ 0x7fffffff); }
__device__ __forceinline__ float ord2f(int k)   { return __int_as_float(k >= 0 ? k : (k ^ 0x7fffffff)); }

__device__ __forceinline__ void edge_sd(const int* ei, int E, int e, int& s, int& d) {
  if (e < E) { s = ei[e]; d = ei[E + e]; } else { s = d = e - E; }  // self-loops
}

__global__ void k_edge_max(const int* __restrict__ ei, int E, int N,
                           const float* __restrict__ asrc, const float* __restrict__ adst,
                           int* __restrict__ mkey) {
  int e = blockIdx.x * blockDim.x + threadIdx.x;
  if (e >= E + N) return;
  int s, d; edge_sd(ei, E, e, s, d);
  float v = asrc[s] + adst[d];
  v = v > 0.f ? v : 0.2f * v;               // leaky_relu(0.2)
  atomicMax(mkey + d, f2ord(v));
}

__global__ void k_edge_expsum(const int* __restrict__ ei, int E, int N,
                              const float* __restrict__ asrc, const float* __restrict__ adst,
                              const int* __restrict__ mkey, float* __restrict__ ssum,
                              float* __restrict__ ex) {
  int e = blockIdx.x * blockDim.x + threadIdx.x;
  if (e >= E + N) return;
  int s, d; edge_sd(ei, E, e, s, d);
  float v = asrc[s] + adst[d];
  v = v > 0.f ? v : 0.2f * v;
  float xv = expf(v - ord2f(mkey[d]));
  ex[e] = xv;
  atomicAdd(ssum + d, xv);
}

// One wave per edge; float4 gather of xp[src], 4 atomic f32 adds per lane.
__global__ void k_edge_agg(const int* __restrict__ ei, int E, int N,
                           const float* __restrict__ ex, const float* __restrict__ ssum,
                           const float* __restrict__ xp, float* __restrict__ agg) {
  long gid = (long)blockIdx.x * blockDim.x + threadIdx.x;
  int e = (int)(gid >> 5), lane = (int)(gid & 31);
  if (e >= E + N) return;
  int s, d; edge_sd(ei, E, e, s, d);
  float alpha = ex[e] / (ssum[d] + 1e-16f);
  float4 v = ((const float4*)(xp + (size_t)s * 128))[lane];
  float* o = agg + (size_t)d * 128 + lane * 4;
  atomicAdd(o + 0, v.x * alpha);
  atomicAdd(o + 1, v.y * alpha);
  atomicAdd(o + 2, v.z * alpha);
  atomicAdd(o + 3, v.w * alpha);
}

__global__ void k_add_row_bias(float* __restrict__ x, const float* __restrict__ b, int N) {
  long i = (long)blockIdx.x * blockDim.x + threadIdx.x;
  if (i >= (long)N * 128) return;
  x[i] += b[i & 127];
}

// ---------------------------------------------------------------------------
// GRU gate combine (+ optional extra addend, optional relu)
// ---------------------------------------------------------------------------
__global__ void k_gru(const float* __restrict__ gi, const float* __restrict__ gh,
                      const float* __restrict__ hprev, const float* __restrict__ extra,
                      float* __restrict__ out, int N, int relu) {
  long i = (long)blockIdx.x * blockDim.x + threadIdx.x;
  if (i >= (long)N * 128) return;
  int n = (int)(i >> 7), c = (int)(i & 127);
  const float* gin = gi + (size_t)n * 384;
  const float* ghn = gh + (size_t)n * 384;
  float r  = 1.f / (1.f + expf(-(gin[c]       + ghn[c])));
  float z  = 1.f / (1.f + expf(-(gin[128 + c] + ghn[128 + c])));
  float nn = tanhf(gin[256 + c] + r * ghn[256 + c]);
  float hv = (1.f - z) * nn + z * hprev[i];
  if (extra) hv += extra[i];
  if (relu)  hv = hv > 0.f ? hv : 0.f;
  out[i] = hv;
}

// ---------------------------------------------------------------------------
// BatchNorm over nodes (training stats, identity affine)
// ---------------------------------------------------------------------------
__global__ void k_bn_stats(const float* __restrict__ x, float* __restrict__ mu,
                           float* __restrict__ ss, int N) {
  int c = threadIdx.x;                       // 128 channels
  int row0 = blockIdx.x * 256;
  int rend = row0 + 256; if (rend > N) rend = N;
  float s = 0.f, q = 0.f;
  for (int r = row0; r < rend; r++) {
    float v = x[(size_t)r * 128 + c];
    s += v; q += v * v;
  }
  atomicAdd(mu + c, s);
  atomicAdd(ss + c, q);
}

__global__ void k_bn_apply(const float* __restrict__ x, const float* __restrict__ mu,
                           const float* __restrict__ ss, float* __restrict__ out, int N) {
  long i = (long)blockIdx.x * blockDim.x + threadIdx.x;
  if (i >= (long)N * 128) return;
  int c = (int)(i & 127);
  float m = mu[c] / (float)N;
  float var = ss[c] / (float)N - m * m;
  out[i] = (x[i] - m) * rsqrtf(var + 1e-5f);
}

// ---------------------------------------------------------------------------
// Host side
// ---------------------------------------------------------------------------
static inline int cdiv(long a, long b) { return (int)((a + b - 1) / b); }

static void gemm(const float* A, const _Float16* Bp, const float* bias, float* C,
                 int N, int K, int M, hipStream_t st) {
  int rowTiles = (N + 15) / 16;
  dim3 grid(M / 64, (rowTiles + 3) / 4);
  if (K == 256) k_wmma_gemm<256><<<grid, dim3(128), 0, st>>>(A, Bp, bias, C, M, rowTiles);
  else          k_wmma_gemm<128><<<grid, dim3(128), 0, st>>>(A, Bp, bias, C, M, rowTiles);
}

extern "C" void kernel_launch(void* const* d_in, const int* in_sizes, int n_in,
                              void* d_out, int out_size, void* d_ws, size_t ws_size,
                              hipStream_t stream) {
  const float* nf     = (const float*)d_in[0];
  const int*   ei     = (const int*)  d_in[1];
  const float* xprev1 = (const float*)d_in[2];
  const float* xprev2 = (const float*)d_in[3];
  const float* ts     = (const float*)d_in[4];
  const float* freq   = (const float*)d_in[5];
  const float* phase  = (const float*)d_in[6];
  const float* mergeW = (const float*)d_in[7];
  const float* mergeB = (const float*)d_in[8];
  const float* g1W    = (const float*)d_in[9];
  const float* g1as   = (const float*)d_in[10];
  const float* g1ad   = (const float*)d_in[11];
  const float* g1b    = (const float*)d_in[12];
  const float* r1Wih  = (const float*)d_in[13];
  const float* r1Whh  = (const float*)d_in[14];
  const float* r1bih  = (const float*)d_in[15];
  const float* r1bhh  = (const float*)d_in[16];
  const float* g2W    = (const float*)d_in[17];
  const float* g2as   = (const float*)d_in[18];
  const float* g2ad   = (const float*)d_in[19];
  const float* g2b    = (const float*)d_in[20];
  const float* r2Wih  = (const float*)d_in[21];
  const float* r2Whh  = (const float*)d_in[22];
  const float* r2bih  = (const float*)d_in[23];
  const float* r2bhh  = (const float*)d_in[24];
  const float* skipW  = (const float*)d_in[25];
  const float* skipB  = (const float*)d_in[26];

  const int N = in_sizes[0] / 128;
  const int E = in_sizes[1] / 2;
  const int Etot = E + N;

  // Workspace layout (floats). Xin aliases gi (disjoint lifetimes).
  float* ws = (float*)d_ws;
  size_t off = 0;
  float* gi   = ws + off; off += (size_t)N * 384;
  float* gh   = ws + off; off += (size_t)N * 384;
  float* x    = ws + off; off += (size_t)N * 128;
  float* xp   = ws + off; off += (size_t)N * 128;
  float* agg  = ws + off; off += (size_t)N * 128;
  float* skip = ws + off; off += (size_t)N * 128;
  float* h2   = ws + off; off += (size_t)N * 128;
  float* asrc = ws + off; off += N;
  float* adst = ws + off; off += N;
  int*   mkey = (int*)(ws + off); off += N;
  float* ssum = ws + off; off += N;
  float* ex   = ws + off; off += Etot;
  float* mu   = ws + off; off += 128;
  float* ss   = ws + off; off += 128;   // adjacent to mu -> single fill of 256
  float* Xin  = gi;                     // alias: Xin dead before gi is written

  // Packed f16 weight fragments (32-byte aligned region)
  off = (off + 7) & ~(size_t)7;
  _Float16* pb = (_Float16*)(ws + off);
  const size_t F128x128 = 4 * 8 * 512;       // (K/32)*(M/16)*512 halves
  const size_t F256x128 = 8 * 8 * 512;
  const size_t F128x384 = 4 * 24 * 512;
  _Float16* pMerge = pb;                 pb += F256x128;
  _Float16* pSkip  = pb;                 pb += F128x128;
  _Float16* pG1    = pb;                 pb += F128x128;
  _Float16* pG2    = pb;                 pb += F128x128;
  _Float16* pR1ih  = pb;                 pb += F128x384;
  _Float16* pR1hh  = pb;                 pb += F128x384;
  _Float16* pR2ih  = pb;                 pb += F128x384;
  _Float16* pR2hh  = pb;                 pb += F128x384;

  const long NC = (long)N * 128;
  const int  INTMIN = (int)0x80000000;

  // 0. Pack weights to f16 fragments
  auto pack = [&](const float* B, _Float16* Bp, int K, int M) {
    int total = (K >> 5) * (M >> 4) * 32;
    k_pack_b<<<cdiv(total, 256), 256, 0, stream>>>(B, Bp, K, M);
  };
  pack(mergeW, pMerge, 256, 128);
  pack(skipW,  pSkip,  128, 128);
  pack(g1W,    pG1,    128, 128);
  pack(g2W,    pG2,    128, 128);
  pack(r1Wih,  pR1ih,  128, 384);
  pack(r1Whh,  pR1hh,  128, 384);
  pack(r2Wih,  pR2ih,  128, 384);
  pack(r2Whh,  pR2hh,  128, 384);

  // 1. Time encode + merge GEMM
  k_time_merge<<<cdiv((long)N * 256, 256), 256, 0, stream>>>(nf, ts, freq, phase, Xin, N);
  gemm(Xin, pMerge, mergeB, x, N, 256, 128, stream);

  // 2. Skip branch (only needs x)
  gemm(x, pSkip, skipB, skip, N, 128, 128, stream);

  // ---- Layer 1: GAT + GRU (+relu) ----
  gemm(x, pG1, nullptr, xp, N, 128, 128, stream);
  k_attn<<<cdiv((long)N * 32, 256), 256, 0, stream>>>(xp, g1as, g1ad, asrc, adst, N);
  k_fill_i32<<<cdiv(N, 256), 256, 0, stream>>>(mkey, INTMIN, N);
  k_fill_f32<<<cdiv(N, 256), 256, 0, stream>>>(ssum, 0.f, N);
  k_fill_f32<<<cdiv(NC, 256), 256, 0, stream>>>(agg, 0.f, NC);
  k_edge_max   <<<cdiv(Etot, 256), 256, 0, stream>>>(ei, E, N, asrc, adst, mkey);
  k_edge_expsum<<<cdiv(Etot, 256), 256, 0, stream>>>(ei, E, N, asrc, adst, mkey, ssum, ex);
  k_edge_agg   <<<cdiv((long)Etot * 32, 256), 256, 0, stream>>>(ei, E, N, ex, ssum, xp, agg);
  k_add_row_bias<<<cdiv(NC, 256), 256, 0, stream>>>(agg, g1b, N);
  gemm(agg,    pR1ih, r1bih, gi, N, 128, 384, stream);
  gemm(xprev1, pR1hh, r1bhh, gh, N, 128, 384, stream);
  float* H1 = (float*)d_out;
  k_gru<<<cdiv(NC, 256), 256, 0, stream>>>(gi, gh, xprev1, nullptr, H1, N, 1);

  // ---- Layer 2: GAT + GRU + skip ----
  gemm(H1, pG2, nullptr, xp, N, 128, 128, stream);
  k_attn<<<cdiv((long)N * 32, 256), 256, 0, stream>>>(xp, g2as, g2ad, asrc, adst, N);
  k_fill_i32<<<cdiv(N, 256), 256, 0, stream>>>(mkey, INTMIN, N);
  k_fill_f32<<<cdiv(N, 256), 256, 0, stream>>>(ssum, 0.f, N);
  k_fill_f32<<<cdiv(NC, 256), 256, 0, stream>>>(agg, 0.f, NC);
  k_edge_max   <<<cdiv(Etot, 256), 256, 0, stream>>>(ei, E, N, asrc, adst, mkey);
  k_edge_expsum<<<cdiv(Etot, 256), 256, 0, stream>>>(ei, E, N, asrc, adst, mkey, ssum, ex);
  k_edge_agg   <<<cdiv((long)Etot * 32, 256), 256, 0, stream>>>(ei, E, N, ex, ssum, xp, agg);
  k_add_row_bias<<<cdiv(NC, 256), 256, 0, stream>>>(agg, g2b, N);
  gemm(agg,    pR2ih, r2bih, gi, N, 128, 384, stream);
  gemm(xprev2, pR2hh, r2bhh, gh, N, 128, 384, stream);
  k_gru<<<cdiv(NC, 256), 256, 0, stream>>>(gi, gh, xprev2, skip, h2, N, 0);

  // ---- BatchNorm -> second half of d_out ----
  k_fill_f32<<<1, 256, 0, stream>>>(mu, 0.f, 256);  // zeros mu and ss
  k_bn_stats<<<cdiv(N, 256), 128, 0, stream>>>(h2, mu, ss, N);
  k_bn_apply<<<cdiv(NC, 256), 256, 0, stream>>>(h2, mu, ss, (float*)d_out + (size_t)N * 128, N);
}
